// SingleheadTransformerDecoderLayer_83786222011041
// MI455X (gfx1250) — matrix operand order repaired and checked by reading, not compiled, not run on hardware
//
#include <hip/hip_runtime.h>
#include <hip/hip_bf16.h>

// ---------------------------------------------------------------------------
// SingleheadTransformerDecoderLayer for MI455X (gfx1250, wave32, WMMA).
// All GEMMs run on v_wmma_f32_16x16x32_bf16 (bf16 operands, f32 accumulate),
// register-blocked 64x32 per wave (4x2 fragments, 8 WMMA per 32-K step).
// V is pre-transposed so every GEMM hits the contiguous-K fast path.
// L=S=2048, N=4, E=1024, FF=4096 -> all dims divisible by 128.
// Workspace assumes ws_size >= ~320 MB (f32 scores dominate at 67 MB).
// ---------------------------------------------------------------------------

typedef __bf16 bf16_t;
typedef __attribute__((ext_vector_type(16))) __bf16 v16bf;
typedef __attribute__((ext_vector_type(8)))  float  v8f;
typedef __attribute__((ext_vector_type(4)))  unsigned int v4u;

__device__ __forceinline__ bf16_t f2bf(float x) {
    unsigned u = __builtin_bit_cast(unsigned, x);
    unsigned r = (u + 0x7FFFu + ((u >> 16) & 1u)) >> 16;   // round-to-nearest-even
    unsigned short s = (unsigned short)r;
    return __builtin_bit_cast(bf16_t, s);
}

// ---------------------------------------------------------------------------
// Batched WMMA GEMM:  C[b](m, j) = sum_k A[b](m,k) * B[b](j,k)
//   A(m,k) at A + b*aBatch + m*lda  + k   (k contiguous, bf16)
//   B(j,k) at B + b*bBatch + j*ldbj + k   (k contiguous, bf16)
// Wave tile: 64x32 (4 M-frags x 2 N-frags); block = 8 waves as 2(M) x 4(N)
// -> block tile 128x128. Grid: (Nout/128, M/128, batch).
// EPI: 0 = bf16 out,  out = (C + bias) * alpha
//      1 = f32 out,   out = C * alpha (+ -1e9 causal mask if col > row)
//      2 = f32 out,   out = resid + C + bias     (residual update, in place)
//      3 = bf16 out,  out = quickgelu(C + bias)  (h * sigmoid(1.702 h))
// Fragment layouts follow CDNA5 ISA 7.12.2 (16-bit A 16x32, B 32x16, f32 C).
// ---------------------------------------------------------------------------
template<int EPI>
__global__ __launch_bounds__(256)
void wmma_gemm_kernel(const bf16_t* __restrict__ A, long lda, long aBatch,
                      const bf16_t* __restrict__ B, long ldbj, long bBatch,
                      const float* __restrict__ bias,
                      const float* __restrict__ resid, long rBatch,
                      void* __restrict__ Cv, long ldc, long cBatch,
                      int K, float alpha, int causal)
{
    const int lane  = threadIdx.x & 31;
    const int wave  = threadIdx.x >> 5;
    const int wm    = wave & 1;                 // 2 waves along M
    const int wn    = wave >> 1;                // 4 waves along N
    const int tileM = blockIdx.y * 128 + wm * 64;
    const int tileN = blockIdx.x * 128 + wn * 32;
    const long b    = blockIdx.z;

    const bf16_t* Ab = A + b * aBatch;
    const bf16_t* Bb = B + b * bBatch;

    // A fragment: lanes 0-15 -> M=lane, K={0..7,16..23};
    // lanes 16-31 -> M=lane-16, K={8..15,24..31}.
    const int am  = lane & 15;
    const int akh = (lane >> 4) & 1;
    const bf16_t* arow = Ab + (long)(tileM + am) * lda + akh * 8;

    // B fragment: n = lane&15; lanes 0-15 -> K={0..7,16..23},
    // lanes 16-31 -> K={8..15,24..31} (same order as A).
    const int bn  = lane & 15;
    const int bko = (lane >> 4) * 8;
    const bf16_t* brow = Bb + (long)(tileN + bn) * ldbj + bko;

    v8f acc[4][2];
    #pragma unroll
    for (int i = 0; i < 4; ++i)
        #pragma unroll
        for (int j = 0; j < 2; ++j)
            acc[i][j] = (v8f){0.f, 0.f, 0.f, 0.f, 0.f, 0.f, 0.f, 0.f};

    for (int k0 = 0; k0 < K; k0 += 32) {
        v16bf af[4];
        #pragma unroll
        for (int i = 0; i < 4; ++i) {
            union { v16bf f; v4u q[2]; } u;
            const bf16_t* p = arow + (long)i * 16 * lda + k0;
            u.q[0] = *(const v4u*)(p);
            u.q[1] = *(const v4u*)(p + 16);
            af[i] = u.f;
        }
        v16bf bfr[2];
        #pragma unroll
        for (int j = 0; j < 2; ++j) {
            union { v16bf f; v4u q[2]; } u;
            const bf16_t* p = brow + (long)j * 16 * ldbj + k0;
            u.q[0] = *(const v4u*)(p);
            u.q[1] = *(const v4u*)(p + 16);
            bfr[j] = u.f;
        }
        if (k0 + 64 < K) {                       // global_prefetch_b8, 2 steps ahead
            __builtin_prefetch(arow + k0 + 64, 0, 3);
            __builtin_prefetch(brow + k0 + 64, 0, 3);
        }
        #pragma unroll
        for (int i = 0; i < 4; ++i)
            #pragma unroll
            for (int j = 0; j < 2; ++j)
                acc[i][j] = __builtin_amdgcn_wmma_f32_16x16x32_bf16(
                    false, af[i], false, bfr[j], (short)0, acc[i][j], false, false);
    }

    // C/D f32 layout: VGPR r, lane l -> M = r + 8*(l>=16), N = l&15.
    const int nloc = lane & 15;
    const int mh   = lane >> 4;

    #pragma unroll
    for (int j = 0; j < 2; ++j) {
        const int col  = tileN + j * 16 + nloc;
        const float bb = bias ? bias[col] : 0.f;
        #pragma unroll
        for (int i = 0; i < 4; ++i) {
            #pragma unroll
            for (int r = 0; r < 8; ++r) {
                const int row = tileM + i * 16 + r + 8 * mh;
                const float c = acc[i][j][r];
                if (EPI == 0) {
                    bf16_t* out = (bf16_t*)Cv + b * cBatch;
                    out[(long)row * ldc + col] = f2bf((c + bb) * alpha);
                } else if (EPI == 1) {
                    float* out = (float*)Cv + b * cBatch;
                    float m = (causal && (col > row)) ? -1e9f : 0.f;
                    out[(long)row * ldc + col] = c * alpha + m;
                } else if (EPI == 2) {
                    float* out = (float*)Cv + b * cBatch;
                    const float* rp = resid + b * rBatch;
                    out[(long)row * ldc + col] = rp[(long)row * ldc + col] + c + bb;
                } else {
                    bf16_t* out = (bf16_t*)Cv + b * cBatch;
                    float h = c + bb;
                    float g = h / (1.f + __expf(-1.702f * h));
                    out[(long)row * ldc + col] = f2bf(g);
                }
            }
        }
    }
}

// ---------------------------------------------------------------------------
// Per-batch transpose: vb(s,n,e) [S,N,E] -> vt(n,e,s) [N,E,S] so the P.V GEMM
// reads V with contiguous K (=s). LDS 32x33 tile kills both-side strides.
// Grid: (E/32, S/32, N), block 256 (32x8).
// ---------------------------------------------------------------------------
__global__ __launch_bounds__(256)
void transpose_v_kernel(const bf16_t* __restrict__ vb, bf16_t* __restrict__ vt,
                        int S, int E, int NB)
{
    __shared__ bf16_t tile[32][33];
    const int n  = blockIdx.z;
    const int e0 = blockIdx.x * 32;
    const int s0 = blockIdx.y * 32;
    const int tx = threadIdx.x & 31;
    const int ty = threadIdx.x >> 5;

    #pragma unroll
    for (int i = 0; i < 32; i += 8) {
        const int s = s0 + ty + i;
        tile[ty + i][tx] = vb[(long)s * NB * E + (long)n * E + e0 + tx];
    }
    __syncthreads();
    #pragma unroll
    for (int i = 0; i < 32; i += 8) {
        const int e = e0 + ty + i;
        vt[(long)n * E * S + (long)e * S + s0 + tx] = tile[tx][ty + i];
    }
}

// ---------------------------------------------------------------------------
// LayerNorm over E=1024 (one block of 256 threads per row), bf16 output.
// ---------------------------------------------------------------------------
__global__ __launch_bounds__(256)
void ln_kernel(const float* __restrict__ x, const float* __restrict__ g,
               const float* __restrict__ bta, bf16_t* __restrict__ out, int E)
{
    const long row = blockIdx.x;
    const float* p = x + row * (long)E;
    const int tid  = threadIdx.x;

    const float4 v4 = *(const float4*)(p + tid * 4);
    float vals[4] = {v4.x, v4.y, v4.z, v4.w};
    float s = 0.f, s2 = 0.f;
    #pragma unroll
    for (int i = 0; i < 4; ++i) { s += vals[i]; s2 += vals[i] * vals[i]; }

    __shared__ float r1[256], r2[256];
    r1[tid] = s; r2[tid] = s2;
    __syncthreads();
    for (int off = 128; off; off >>= 1) {
        if (tid < off) { r1[tid] += r1[tid + off]; r2[tid] += r2[tid + off]; }
        __syncthreads();
    }
    const float mean = r1[0] / E;
    const float var  = r2[0] / E - mean * mean;
    const float rstd = rsqrtf(var + 1e-5f);

    bf16_t* op = out + row * (long)E + tid * 4;
    const float* gp = g + tid * 4;
    const float* bp = bta + tid * 4;
    #pragma unroll
    for (int i = 0; i < 4; ++i)
        op[i] = f2bf((vals[i] - mean) * rstd * gp[i] + bp[i]);
}

// ---------------------------------------------------------------------------
// Row softmax over S=2048 (one block per row), f32 scores -> bf16 probs.
// ---------------------------------------------------------------------------
__global__ __launch_bounds__(256)
void softmax_kernel(const float* __restrict__ scores, bf16_t* __restrict__ probs, int S)
{
    const long row = blockIdx.x;
    const float* p = scores + row * (long)S;
    const int tid  = threadIdx.x;

    float v[8];
    #pragma unroll
    for (int i = 0; i < 8; ++i) v[i] = p[tid + i * 256];

    float m = v[0];
    #pragma unroll
    for (int i = 1; i < 8; ++i) m = fmaxf(m, v[i]);

    __shared__ float red[256];
    red[tid] = m;
    __syncthreads();
    for (int off = 128; off; off >>= 1) {
        if (tid < off) red[tid] = fmaxf(red[tid], red[tid + off]);
        __syncthreads();
    }
    m = red[0];
    __syncthreads();

    float s = 0.f;
    #pragma unroll
    for (int i = 0; i < 8; ++i) { v[i] = __expf(v[i] - m); s += v[i]; }
    red[tid] = s;
    __syncthreads();
    for (int off = 128; off; off >>= 1) {
        if (tid < off) red[tid] += red[tid + off];
        __syncthreads();
    }
    const float inv = 1.f / red[0];

    bf16_t* op = probs + row * (long)S;
    #pragma unroll
    for (int i = 0; i < 8; ++i) op[tid + i * 256] = f2bf(v[i] * inv);
}

// ---------------------------------------------------------------------------
// f32 -> bf16 elementwise conversion (weights / memory activations).
// ---------------------------------------------------------------------------
__global__ void cvt_kernel(const float* __restrict__ in, bf16_t* __restrict__ out, long n)
{
    long i = (long)blockIdx.x * blockDim.x + threadIdx.x;
    const long stride = (long)gridDim.x * blockDim.x;
    for (; i < n; i += stride) out[i] = f2bf(in[i]);
}

// ---------------------------------------------------------------------------

extern "C" void kernel_launch(void* const* d_in, const int* in_sizes, int n_in,
                              void* d_out, int out_size, void* d_ws, size_t ws_size,
                              hipStream_t stream)
{
    constexpr int  L = 2048, S = 2048, NBATCH = 4, E = 1024, FF = 4096;
    constexpr long LN_ = (long)L * NBATCH;             // rows of the residual stream

    const float* tgt   = (const float*)d_in[0];
    const float* mem   = (const float*)d_in[1];
    // d_in[2] == tgt_mask (causal, reproduced analytically in the epilogue)
    const float* sa_qw = (const float*)d_in[3];
    const float* sa_kw = (const float*)d_in[4];
    const float* sa_vw = (const float*)d_in[5];
    const float* sa_b  = (const float*)d_in[6];
    const float* sa_ow = (const float*)d_in[7];
    const float* sa_ob = (const float*)d_in[8];
    const float* ca_qw = (const float*)d_in[9];
    const float* ca_kw = (const float*)d_in[10];
    const float* ca_vw = (const float*)d_in[11];
    const float* ca_b  = (const float*)d_in[12];
    const float* ca_ow = (const float*)d_in[13];
    const float* ca_ob = (const float*)d_in[14];
    const float* ln1g  = (const float*)d_in[15];
    const float* ln1b  = (const float*)d_in[16];
    const float* ln2g  = (const float*)d_in[17];
    const float* ln2b  = (const float*)d_in[18];
    const float* ln3g  = (const float*)d_in[19];
    const float* ln3b  = (const float*)d_in[20];
    const float* w1    = (const float*)d_in[21];
    const float* b1    = (const float*)d_in[22];
    const float* w2    = (const float*)d_in[23];
    const float* b2    = (const float*)d_in[24];

    float* x = (float*)d_out;                          // residual stream [L,N,E] f32

    // -------- workspace bump allocator (256B aligned) ----------------------
    char*  base = (char*)d_ws;
    size_t off  = 0;
    auto alloc = [&](size_t bytes) -> void* {
        void* p = base + off;
        off = (off + bytes + 255) & ~(size_t)255;
        return p;
    };
    bf16_t* hb   = (bf16_t*)alloc((size_t)LN_ * E * 2);        // LN output
    bf16_t* memb = (bf16_t*)alloc((size_t)S * NBATCH * E * 2); // memory in bf16
    bf16_t* qb   = (bf16_t*)alloc((size_t)LN_ * E * 2);
    bf16_t* kb   = (bf16_t*)alloc((size_t)S * NBATCH * E * 2);
    bf16_t* vb   = (bf16_t*)alloc((size_t)S * NBATCH * E * 2);
    bf16_t* vt   = (bf16_t*)alloc((size_t)NBATCH * E * S * 2); // V transposed
    float*  sc   = (float* )alloc((size_t)NBATCH * L * S * 4); // scores f32
    bf16_t* pb   = (bf16_t*)alloc((size_t)NBATCH * L * S * 2); // probs bf16
    bf16_t* ao   = (bf16_t*)alloc((size_t)LN_ * E * 2);        // attn out bf16
    bf16_t* hid  = (bf16_t*)alloc((size_t)LN_ * FF * 2);       // MLP hidden bf16
    bf16_t* wqb  = (bf16_t*)alloc((size_t)E * E * 2);
    bf16_t* wkb  = (bf16_t*)alloc((size_t)E * E * 2);
    bf16_t* wvb  = (bf16_t*)alloc((size_t)E * E * 2);
    bf16_t* wob  = (bf16_t*)alloc((size_t)E * E * 2);
    bf16_t* w1b  = (bf16_t*)alloc((size_t)FF * E * 2);
    bf16_t* w2b  = (bf16_t*)alloc((size_t)E * FF * 2);
    (void)ws_size; (void)in_sizes; (void)n_in; (void)out_size;

    auto cvt = [&](const float* in, bf16_t* out, long n) {
        long blocks = (n + 255) / 256; if (blocks > 4096) blocks = 4096;
        cvt_kernel<<<dim3((unsigned)blocks), dim3(256), 0, stream>>>(in, out, n);
    };

    const float scale = 0.03125f;                     // E^-0.5 = 1/32
    const dim3 blk(256);
    const dim3 gProj (E  / 128, (unsigned)(LN_ / 128), 1);      // [8192 x 1024]
    const dim3 gM1   (FF / 128, (unsigned)(LN_ / 128), 1);      // [8192 x 4096]
    const dim3 gScore(S  / 128, L / 128, NBATCH);               // [2048 x 2048] x N
    const dim3 gPV   (E  / 128, L / 128, NBATCH);               // [2048 x 1024] x N
    const dim3 gTr   (E  / 32,  S / 32,  NBATCH);

    // residual stream starts as tgt
    hipMemcpyAsync(x, tgt, (size_t)LN_ * E * 4, hipMemcpyDeviceToDevice, stream);

    // one-time conversions
    cvt(mem, memb, (long)S * NBATCH * E);
    cvt(w1, w1b, (long)FF * E);
    cvt(w2, w2b, (long)E * FF);

    // ================= self-attention (pre-norm, causal) ===================
    cvt(sa_qw, wqb, (long)E * E);
    cvt(sa_kw, wkb, (long)E * E);
    cvt(sa_vw, wvb, (long)E * E);
    cvt(sa_ow, wob, (long)E * E);

    ln_kernel<<<(unsigned)LN_, blk, 0, stream>>>(x, ln1g, ln1b, hb, E);

    // q = (h @ qw^T + b[0:E]) * scale ; k,v analogous (scale folded into q)
    wmma_gemm_kernel<0><<<gProj, blk, 0, stream>>>(
        hb, E, 0, wqb, E, 0, sa_b, nullptr, 0, qb, E, 0, E, scale, 0);
    wmma_gemm_kernel<0><<<gProj, blk, 0, stream>>>(
        hb, E, 0, wkb, E, 0, sa_b + E, nullptr, 0, kb, E, 0, E, 1.f, 0);
    wmma_gemm_kernel<0><<<gProj, blk, 0, stream>>>(
        hb, E, 0, wvb, E, 0, sa_b + 2 * E, nullptr, 0, vb, E, 0, E, 1.f, 0);

    transpose_v_kernel<<<gTr, blk, 0, stream>>>(vb, vt, S, E, NBATCH);

    // scores[n,l,s] = q(l,n,:) . k(s,n,:)  (+ causal mask)
    wmma_gemm_kernel<1><<<gScore, blk, 0, stream>>>(
        qb, (long)NBATCH * E, E, kb, (long)NBATCH * E, E,
        nullptr, nullptr, 0, sc, S, (long)L * S, E, 1.f, 1);

    softmax_kernel<<<(unsigned)(NBATCH * L), blk, 0, stream>>>(sc, pb, S);

    // ao[l,n,e] = sum_s P[n,l,s] * Vt[n,e,s]  (contiguous-K via transpose)
    wmma_gemm_kernel<0><<<gPV, blk, 0, stream>>>(
        pb, S, (long)L * S, vt, S, (long)E * S,
        nullptr, nullptr, 0, ao, (long)NBATCH * E, E, S, 1.f, 0);

    // x += ao @ ow^T + ob  (residual epilogue, in place on d_out)
    wmma_gemm_kernel<2><<<gProj, blk, 0, stream>>>(
        ao, E, 0, wob, E, 0, sa_ob, x, 0, x, E, 0, E, 1.f, 0);

    // ================= cross-attention (pre-norm, no mask) =================
    cvt(ca_qw, wqb, (long)E * E);
    cvt(ca_kw, wkb, (long)E * E);
    cvt(ca_vw, wvb, (long)E * E);
    cvt(ca_ow, wob, (long)E * E);

    ln_kernel<<<(unsigned)LN_, blk, 0, stream>>>(x, ln2g, ln2b, hb, E);

    wmma_gemm_kernel<0><<<gProj, blk, 0, stream>>>(
        hb, E, 0, wqb, E, 0, ca_b, nullptr, 0, qb, E, 0, E, scale, 0);
    // k, v from encoder memory
    wmma_gemm_kernel<0><<<gProj, blk, 0, stream>>>(
        memb, E, 0, wkb, E, 0, ca_b + E, nullptr, 0, kb, E, 0, E, 1.f, 0);
    wmma_gemm_kernel<0><<<gProj, blk, 0, stream>>>(
        memb, E, 0, wvb, E, 0, ca_b + 2 * E, nullptr, 0, vb, E, 0, E, 1.f, 0);

    transpose_v_kernel<<<gTr, blk, 0, stream>>>(vb, vt, S, E, NBATCH);

    wmma_gemm_kernel<1><<<gScore, blk, 0, stream>>>(
        qb, (long)NBATCH * E, E, kb, (long)NBATCH * E, E,
        nullptr, nullptr, 0, sc, S, (long)L * S, E, 1.f, 0);

    softmax_kernel<<<(unsigned)(NBATCH * L), blk, 0, stream>>>(sc, pb, S);

    wmma_gemm_kernel<0><<<gPV, blk, 0, stream>>>(
        pb, S, (long)L * S, vt, S, (long)E * S,
        nullptr, nullptr, 0, ao, (long)NBATCH * E, E, S, 1.f, 0);

    wmma_gemm_kernel<2><<<gProj, blk, 0, stream>>>(
        ao, E, 0, wob, E, 0, ca_ob, x, 0, x, E, 0, E, 1.f, 0);

    // ================= MLP with QuickGELU ==================================
    ln_kernel<<<(unsigned)LN_, blk, 0, stream>>>(x, ln3g, ln3b, hb, E);

    // hid = quickgelu(h @ w1^T + b1)   -> bf16
    wmma_gemm_kernel<3><<<gM1, blk, 0, stream>>>(
        hb, E, 0, w1b, E, 0, b1, nullptr, 0, hid, FF, 0, E, 1.f, 0);

    // x += hid @ w2^T + b2
    wmma_gemm_kernel<2><<<gProj, blk, 0, stream>>>(
        hid, FF, 0, w2b, FF, 0, b2, x, 0, x, E, 0, FF, 1.f, 0);
}
